// GCN_60241211294442
// MI455X (gfx1250) — matrix operand order repaired and checked by reading, not compiled, not run on hardware
//
#include <hip/hip_runtime.h>

// GCN forward on MI455X (gfx1250), bf16 WMMA path, split-K for occupancy.
// Roofline: 2 passes over 256MB adj + 1 rowsum pass ~ 768MB @ 23.3TB/s ~ 33us;
// bf16 WMMA compute (~26 GFLOP) is far below that, so the goal is to stay
// HBM-bound: coalesced adj streaming, v_perm_b32 packing (3 VALU / bf16 pair),
// 512-block grids (~4 waves/SIMD on ~256 WGPs), deterministic split-K partials.

typedef __attribute__((ext_vector_type(16))) __bf16       v16bf;
typedef __attribute__((ext_vector_type(8)))  float        v8f;
typedef __attribute__((ext_vector_type(4)))  unsigned int u32x4;
typedef __attribute__((ext_vector_type(4)))  float        f32x4;

#define DEVINL __device__ __forceinline__

static constexpr int NN    = 8192;  // nodes
static constexpr int EMB   = 256;   // input embedding
static constexpr int HID   = 128;   // hidden
static constexpr int ODIM  = 64;    // output
static constexpr int SPLIT = 8;     // K-split for the big GEMMs

union Frag {
  u32x4 q[2];
  v16bf v;
};

// f32 -> bf16 with round-half-up (bias 2^-9 ulp-relative; << bf16 quantization)
DEVINL unsigned int f2bf(float f) {
  return (__builtin_bit_cast(unsigned int, f) + 0x8000u) >> 16;
}
// pack two f32 -> {bf16(hi), bf16(lo)} in one v_perm_b32 (+2 adds)
DEVINL unsigned int pack2bf(float lo, float hi) {
  unsigned int ul = __builtin_bit_cast(unsigned int, lo) + 0x8000u;
  unsigned int uh = __builtin_bit_cast(unsigned int, hi) + 0x8000u;
  return __builtin_amdgcn_perm(uh, ul, 0x07060302u);
}

// ------------------------------------------------------------------
// d[i] = rsqrt( sum_j adj[i][j] )   -- one block per row, coalesced
// ------------------------------------------------------------------
__global__ __launch_bounds__(256) void k_rowsum(const float* __restrict__ adj,
                                                float* __restrict__ dvec) {
  const int row = blockIdx.x;
  const float* p = adj + (size_t)row * NN;
  float s = 0.f;
  for (int c = threadIdx.x * 4; c < NN; c += 1024) {
    f32x4 v = *(const f32x4*)(p + c);
    s += (v.x + v.y) + (v.z + v.w);
  }
  __shared__ float red[256];
  red[threadIdx.x] = s;
  __syncthreads();
  for (int off = 128; off > 0; off >>= 1) {
    if ((int)threadIdx.x < off) red[threadIdx.x] += red[threadIdx.x + off];
    __syncthreads();
  }
  if (threadIdx.x == 0) dvec[row] = rsqrtf(red[0]);
}

// ------------------------------------------------------------------
// Small GEMM (transposed output):
//   GT[n, j] = d[j] * sum_k W[k, n] * B[j, k]        (bf16 out, [NM, NN])
// A-matrix = W^T staged to LDS as bf16 (pair-packed b32 stores);
// B rows read directly per lane (contiguous K runs).
// ------------------------------------------------------------------
template<int KD, int NM, bool BF32>
__global__ __launch_bounds__(256) void k_small_gemm(
    const float* __restrict__ Bf,            // [NN, KD] f32   (when BF32)
    const unsigned short* __restrict__ Bh,   // [NN, KD] bf16  (when !BF32)
    const float* __restrict__ W,             // [KD, NM] f32 row-major
    const float* __restrict__ dvec,          // [NN]
    unsigned short* __restrict__ GT) {       // [NM, NN] bf16 out
  constexpr int NSTRIPS = NM / 16;
  constexpr int WPS     = 8 / NSTRIPS;   // waves per n-strip
  constexpr int JPW     = 128 / WPS;     // j columns per wave
  constexpr int F       = JPW / 16;      // 16x16 frags per wave

  __shared__ __attribute__((aligned(16))) unsigned short WT[NM][KD + 8];

  const int tid    = threadIdx.x;
  const int wave   = tid >> 5;
  const int lane   = tid & 31;
  const int jblk   = blockIdx.x * 128;
  const int nstrip = wave % NSTRIPS;
  const int jw     = (wave / NSTRIPS) * JPW;
  const int lhalf  = lane >> 4;
  const int nA     = nstrip * 16 + (lane & 15);

  // Stage W^T into LDS as bf16, two k's packed per b32 store
  for (int p = tid; p < (KD * NM) / 2; p += 256) {
    const int n  = p % NM;
    const int k2 = (p / NM) * 2;
    *(unsigned int*)&WT[n][k2] =
        pack2bf(W[(size_t)k2 * NM + n], W[(size_t)(k2 + 1) * NM + n]);
  }
  __syncthreads();

  v8f acc[F];
#pragma unroll
  for (int t = 0; t < F; ++t)
#pragma unroll
    for (int e = 0; e < 8; ++e) acc[t][e] = 0.0f;

  for (int k0 = 0; k0 < KD; k0 += 32) {
    // A frag: lane<16 -> K k0..k0+15 ; lane>=16 -> K k0+16..k0+31
    Frag a;
    const int kb = k0 + lhalf * 16;
    a.q[0] = *(const u32x4*)&WT[nA][kb];
    a.q[1] = *(const u32x4*)&WT[nA][kb + 8];

    const int kb2 = k0 + lhalf * 16;  // B frag: 16 contiguous K per lane
#pragma unroll
    for (int t = 0; t < F; ++t) {
      const int j = jblk + jw + t * 16 + (lane & 15);
      Frag b;
      if constexpr (BF32) {
        const f32x4* bp = (const f32x4*)(Bf + (size_t)j * KD + kb2);
        f32x4 f0 = bp[0], f1 = bp[1], f2 = bp[2], f3 = bp[3];
        u32x4 q0, q1;
        q0[0] = pack2bf(f0.x, f0.y); q0[1] = pack2bf(f0.z, f0.w);
        q0[2] = pack2bf(f1.x, f1.y); q0[3] = pack2bf(f1.z, f1.w);
        q1[0] = pack2bf(f2.x, f2.y); q1[1] = pack2bf(f2.z, f2.w);
        q1[2] = pack2bf(f3.x, f3.y); q1[3] = pack2bf(f3.z, f3.w);
        b.q[0] = q0; b.q[1] = q1;
      } else {
        const u32x4* bp = (const u32x4*)(Bh + (size_t)j * KD + kb2);
        b.q[0] = bp[0];
        b.q[1] = bp[1];
      }
      acc[t] = __builtin_amdgcn_wmma_f32_16x16x32_bf16(
          false, a.v, false, b.v, (short)0, acc[t], false, false);
    }
  }

  // Epilogue: scale by d[j], pack 8 consecutive j as bf16 -> one 16B store
#pragma unroll
  for (int t = 0; t < F; ++t) {
    const int jb = jblk + jw + t * 16 + lhalf * 8;
    f32x4 d0 = *(const f32x4*)(dvec + jb);
    f32x4 d1 = *(const f32x4*)(dvec + jb + 4);
    u32x4 o;
    o[0] = pack2bf(acc[t][0] * d0.x, acc[t][1] * d0.y);
    o[1] = pack2bf(acc[t][2] * d0.z, acc[t][3] * d0.w);
    o[2] = pack2bf(acc[t][4] * d1.x, acc[t][5] * d1.y);
    o[3] = pack2bf(acc[t][6] * d1.z, acc[t][7] * d1.w);
    *(u32x4*)(GT + (size_t)nA * NN + jb) = o;
  }
}

// ------------------------------------------------------------------
// Big GEMM against adj (transposed domain), split-K:
//   Cpart[split][n, i] = sum_{j in split} GT[n, j] * adj[j, i]   (f32)
// adj tile [32j x 128i] f32 -> LDS bf16 pair-packed & transposed (pitch 20),
// with the next tile software-prefetched into registers during WMMA.
// ------------------------------------------------------------------
template<int NM>
__global__ __launch_bounds__(256) void k_big_gemm(
    const unsigned short* __restrict__ GT,   // [NM, NN] bf16
    const float* __restrict__ adj,           // [NN, NN] f32, indexed (j, i)
    float* __restrict__ Cpart) {             // [SPLIT, NM, NN] f32
  constexpr int NSTRIPS = NM / 16;
  constexpr int WPS     = 8 / NSTRIPS;
  constexpr int IPW     = 128 / WPS;
  constexpr int F       = IPW / 16;
  constexpr int PITCH   = 20;  // dwords per i-row: 16 packed-bf16 pairs + pad
  constexpr int JSPAN   = NN / SPLIT;

  __shared__ __attribute__((aligned(16))) unsigned int LT[128 * PITCH];

  const int tid    = threadIdx.x;
  const int wave   = tid >> 5;
  const int lane   = tid & 31;
  const int iblk   = blockIdx.x * 128;
  const int jbase  = blockIdx.y * JSPAN;
  const int nstrip = wave % NSTRIPS;
  const int iw     = (wave / NSTRIPS) * IPW;
  const int lhalf  = lane >> 4;
  const int nA     = nstrip * 16 + (lane & 15);

  const int c  = tid & 15;   // column group within i-tile
  const int jp = tid >> 4;   // j-pair 0..15

  v8f acc[F];
#pragma unroll
  for (int t = 0; t < F; ++t)
#pragma unroll
    for (int e = 0; e < 8; ++e) acc[t][e] = 0.0f;

  const unsigned short* gp = GT + (size_t)nA * NN;

  float pa[8], pb[8];
  auto load_tile = [&](int j0) {
    const float* r0 = adj + (size_t)(j0 + 2 * jp) * NN + iblk;
    const float* r1 = r0 + NN;
#pragma unroll
    for (int q = 0; q < 8; ++q) {   // coalesced across lanes
      const int ic = c + 16 * q;
      pa[q] = r0[ic];
      pb[q] = r1[ic];
    }
  };

  load_tile(jbase);
  for (int j0 = jbase; j0 < jbase + JSPAN; j0 += 32) {
    __syncthreads();  // previous tile fully consumed
#pragma unroll
    for (int q = 0; q < 8; ++q) {
      const int ic = c + 16 * q;
      LT[ic * PITCH + jp] = pack2bf(pa[q], pb[q]);  // conflict-free, 1 perm
    }
    __syncthreads();

    if (j0 + 32 < jbase + JSPAN) load_tile(j0 + 32);  // prefetch next tile

    // A frag from GT (L2-resident, 16B aligned runs)
    Frag a;
    const int kb = j0 + lhalf * 16;
    a.q[0] = *(const u32x4*)(gp + kb);
    a.q[1] = *(const u32x4*)(gp + kb + 8);

#pragma unroll
    for (int t = 0; t < F; ++t) {
      const int il = iw + t * 16 + (lane & 15);
      const unsigned int* lp = LT + il * PITCH + lhalf * 8;
      Frag b;
      b.q[0] = *(const u32x4*)lp;        // ds_load_b128
      b.q[1] = *(const u32x4*)(lp + 4);  // ds_load_b128
      acc[t] = __builtin_amdgcn_wmma_f32_16x16x32_bf16(
          false, a.v, false, b.v, (short)0, acc[t], false, false);
    }
  }

  // Store f32 partials: lane owns row n, 8 consecutive i -> two b128 stores
#pragma unroll
  for (int t = 0; t < F; ++t) {
    const int ib = iblk + iw + t * 16 + lhalf * 8;
    float* cp = Cpart + ((size_t)blockIdx.y * NM + nA) * NN + ib;
    f32x4 lo = {acc[t][0], acc[t][1], acc[t][2], acc[t][3]};
    f32x4 hi = {acc[t][4], acc[t][5], acc[t][6], acc[t][7]};
    *(f32x4*)cp = lo;
    *(f32x4*)(cp + 4) = hi;
  }
}

// ------------------------------------------------------------------
// Reduce split-K partials + epilogue (deterministic, no atomics):
//   FIRST : h[i,n]  = relu(d[i]*sum_s C[s][n][i] + b1[n])  bf16 [NN,HID]
//   !FIRST: out[i,n] =      d[i]*sum_s C[s][n][i] + b2[n]  f32  [NN,ODIM]
// ------------------------------------------------------------------
template<int NM, bool FIRST>
__global__ __launch_bounds__(256) void k_finish(
    const float* __restrict__ Cpart,   // [SPLIT, NM, NN]
    const float* __restrict__ dvec,
    const float* __restrict__ bias,    // [NM]
    unsigned short* __restrict__ hRM,  // [NN, HID] bf16 (FIRST)
    float* __restrict__ outF) {        // [NN, ODIM] f32 (!FIRST)
  const int i = blockIdx.x * 256 + threadIdx.x;
  const int n = blockIdx.y;
  float s = 0.f;
#pragma unroll
  for (int sp = 0; sp < SPLIT; ++sp)
    s += Cpart[((size_t)sp * NM + n) * NN + i];   // coalesced
  float v = s * dvec[i] + bias[n];
  if constexpr (FIRST) {
    v = fmaxf(v, 0.0f);
    hRM[(size_t)i * HID + n] = (unsigned short)f2bf(v);
  } else {
    outF[(size_t)i * ODIM + n] = v;
  }
}

// ------------------------------------------------------------------
extern "C" void kernel_launch(void* const* d_in, const int* in_sizes, int n_in,
                              void* d_out, int out_size, void* d_ws, size_t ws_size,
                              hipStream_t stream) {
  (void)in_sizes; (void)n_in; (void)out_size; (void)ws_size;
  const float* x   = (const float*)d_in[0];  // [8192, 256]
  const float* adj = (const float*)d_in[1];  // [8192, 8192]
  const float* W1  = (const float*)d_in[2];  // [256, 128]
  const float* b1  = (const float*)d_in[3];  // [128]
  const float* W2  = (const float*)d_in[4];  // [128, 64]
  const float* b2  = (const float*)d_in[5];  // [64]
  float* out = (float*)d_out;                // [8192, 64]

  char* ws = (char*)d_ws;
  float*          dvec  = (float*)ws;                                    // 32 KB
  unsigned short* PT    = (unsigned short*)(ws + 32768);                 // 2 MB [128,8192]
  unsigned short* hB    = (unsigned short*)(ws + 32768 + 2097152);       // 2 MB [8192,128]
  unsigned short* QT    = (unsigned short*)(ws + 32768 + 2 * 2097152);   // 1 MB [64,8192]
  float*          Cpart = (float*)(ws + 32768 + 5242880);                // 32 MB max

  // 1) degree scaling vector
  k_rowsum<<<NN, 256, 0, stream>>>(adj, dvec);
  // 2) P'^T[n,j] = d[j] * (x @ W1)^T
  k_small_gemm<EMB, HID, true><<<NN / 128, 256, 0, stream>>>(x, nullptr, W1, dvec, PT);
  // 3) Cpart = split-K (P'^T @ adj);  h = relu(d[i]*sum + b1)
  k_big_gemm<HID><<<dim3(NN / 128, SPLIT), 256, 0, stream>>>(PT, adj, Cpart);
  k_finish<HID, true><<<dim3(NN / 256, HID), 256, 0, stream>>>(Cpart, dvec, b1, hB, nullptr);
  // 4) Q'^T[n,j] = d[j] * (h @ W2)^T
  k_small_gemm<HID, ODIM, false><<<NN / 128, 256, 0, stream>>>(nullptr, hB, W2, dvec, QT);
  // 5) Cpart = split-K (Q'^T @ adj);  out = d[i]*sum + b2
  k_big_gemm<ODIM><<<dim3(NN / 128, SPLIT), 256, 0, stream>>>(QT, adj, Cpart);
  k_finish<ODIM, false><<<dim3(NN / 256, ODIM), 256, 0, stream>>>(Cpart, dvec, b2, nullptr, out);
}
